// Barrier_Net_81080392614147
// MI455X (gfx1250) — compile-verified, matrix operand order checked by reading, not run on hardware
//
#include <hip/hip_runtime.h>
#include <hip/hip_bf16.h>
#include <math.h>

typedef _Float16 half_t;
typedef __attribute__((ext_vector_type(16))) _Float16 v16h;
typedef __attribute__((ext_vector_type(8)))  _Float16 v8h;
typedef __attribute__((ext_vector_type(8)))  float    v8f;

// ---------------------------------------------------------------------------
// Workspace layout (half offsets for weights, float offsets for biases).
// Weight tiles: 512 halves per 16x16x32 B-tile, ordered [ntile][ktile],
// within a tile: [lane 0..31][16 contiguous halves] in WMMA B lane layout.
// ---------------------------------------------------------------------------
enum : int {
  W_PHIN0 = 0,     W_PHIN1 = 2048,  W_PHIN2 = 6144,
  W_RHON0 = 7168,  W_RHON1 = 9216,
  W_PHIO0 = 10240, W_PHIO1 = 12288, W_PHIO2 = 16384,
  W_RHOO0 = 17408, W_RHOO1 = 19456,
  W_PSI0  = 20480, W_PSI1  = 22528, W_PSI2  = 26624,
  W_TOTAL_HALVES = 27648,                 // 54 tiles * 512
  B_PHIN0 = 0,   B_PHIN1 = 64,  B_PHIN2 = 128,
  B_RHON0 = 144, B_RHON1 = 208,
  B_PHIO0 = 224, B_PHIO1 = 288, B_PHIO2 = 352,
  B_RHOO0 = 368, B_RHOO1 = 432,
  B_PSI0  = 448, B_PSI1  = 512, B_PSI2  = 576,
  BIAS_BYTE_OFF = W_TOTAL_HALVES * 2      // 55296, 16B aligned
};

// ---------------------------------------------------------------------------
// Weight pre-swizzle: fp32 (dout,din) row-major -> f16 WMMA B tiles.
// K-map per lane (matches 16-bit A/B VGPR striping in the CDNA5 ISA):
//   lane L holds N = ntile*16 + (L&15), kh = (L>=16)*8
//   element e -> K = ktile*32 + kh + e + (e>=8 ? 8 : 0)
// ---------------------------------------------------------------------------
__global__ void swizzle_w_kernel(const float* __restrict__ W, int din, int dout,
                                 half_t* __restrict__ dst, int KT, int NT) {
  int idx = blockIdx.x * blockDim.x + threadIdx.x;
  int total = NT * KT * 512;
  if (idx >= total) return;
  int e    = idx & 15;
  int lane = (idx >> 4) & 31;
  int tile = idx >> 9;
  int kt   = tile % KT;
  int nt   = tile / KT;
  int kh   = (lane >> 4) << 3;
  int k    = kt * 32 + kh + e + ((e >= 8) ? 8 : 0);
  int n    = nt * 16 + (lane & 15);
  float v  = (k < din && n < dout) ? W[n * din + k] : 0.0f;
  dst[idx] = (half_t)v;
}

__global__ void pad_bias_kernel(const float* __restrict__ b, int dout,
                                float* __restrict__ dst, int npad) {
  int i = blockIdx.x * blockDim.x + threadIdx.x;
  if (i < npad) dst[i] = (i < dout) ? b[i] : 0.0f;
}

// ---------------------------------------------------------------------------
// A-operand load from LDS: activations stored row-major [m][k] f16.
// lane L: m = L&15, kh = (L>=16)*8; halves K = {kc*32+kh..+7, kc*32+16+kh..+7}
// ---------------------------------------------------------------------------
__device__ __forceinline__ v16h lds_load_a(const half_t* act, int stride,
                                           int kc, int lane) {
  int m  = lane & 15;
  int kh = (lane >> 4) << 3;
  const half_t* p = act + m * stride + kc * 32 + kh;
  v8h lo = *(const v8h*)(p);
  v8h hi = *(const v8h*)(p + 16);
  v16h r;
#pragma unroll
  for (int i = 0; i < 8; ++i) { r[i] = lo[i]; r[i + 8] = hi[i]; }
  return r;
}

// Hidden layer: M=16 rows, N=64 out (4 N-tiles), KT k-chunks, bias+ReLU,
// result written to f16 LDS buffer (row stride 64). Safe in-place (A is
// fully loaded to VGPRs before any store).
template <int KT>
__device__ __forceinline__ void mlp_layer64(const half_t* actin, int instride,
                                            const half_t* __restrict__ W,
                                            const float* __restrict__ bias,
                                            half_t* actout, int lane) {
  v16h a[KT];
#pragma unroll
  for (int kc = 0; kc < KT; ++kc) a[kc] = lds_load_a(actin, instride, kc, lane);
  int n  = lane & 15;
  int mh = (lane >> 4) << 3;
#pragma unroll
  for (int t = 0; t < 4; ++t) {
    v8f acc = {};
#pragma unroll
    for (int kc = 0; kc < KT; ++kc) {
      v16h b = *(const v16h*)(W + (size_t)(t * KT + kc) * 512 + lane * 16);
      acc = __builtin_amdgcn_wmma_f32_16x16x32_f16(false, a[kc], false, b,
                                                   (short)0, acc, false, false);
    }
    float bv = bias[t * 16 + n];
#pragma unroll
    for (int r = 0; r < 8; ++r) {
      float v = acc[r] + bv;
      v = v > 0.0f ? v : 0.0f;
      actout[(r + mh) * 64 + t * 16 + n] = (half_t)v;
    }
  }
}

// Output layer: single N-tile (16 cols), returns raw f32 accumulators
// (bias handled by caller; last layers are linear).
template <int KT>
__device__ __forceinline__ v8f mlp_layer_out(const half_t* actin, int instride,
                                             const half_t* __restrict__ W,
                                             int lane) {
  v8f acc = {};
#pragma unroll
  for (int kc = 0; kc < KT; ++kc) {
    v16h a = lds_load_a(actin, instride, kc, lane);
    v16h b = *(const v16h*)(W + (size_t)kc * 512 + lane * 16);
    acc = __builtin_amdgcn_wmma_f32_16x16x32_f16(false, a, false, b,
                                                 (short)0, acc, false, false);
  }
  return acc;
}

// One sample's deep-set encoder branch: 16 elements x [din->64->64->16],
// column-summed over the 16 elements -> 16-dim set feature (f16 into sdst).
__device__ __forceinline__ void phi_pipe(const float* __restrict__ xrow, int obs,
                                         const half_t* W0, const float* B0,
                                         const half_t* W1, const float* B1,
                                         const half_t* W2, const float* B2,
                                         half_t* actw, half_t* sdst, int lane) {
  // zero the 16x64 f16 staging buffer (pads K of layer0 to 32)
#pragma unroll
  for (int i = 0; i < 4; ++i)
    ((float4*)actw)[lane * 4 + i] = make_float4(0.f, 0.f, 0.f, 0.f);
  if (!obs) {              // neighbors: 16 rows x 4 features at xrow[5..68]
    int f0 = lane * 2, f1 = lane * 2 + 1;
    actw[(f0 >> 2) * 64 + (f0 & 3)] = (half_t)xrow[5 + f0];
    actw[(f1 >> 2) * 64 + (f1 & 3)] = (half_t)xrow[5 + f1];
  } else {                 // obstacles: 16 rows x 2 features at xrow[69..100]
    actw[(lane >> 1) * 64 + (lane & 1)] = (half_t)xrow[69 + lane];
  }
  mlp_layer64<1>(actw, 64, W0, B0, actw, lane);
  mlp_layer64<2>(actw, 64, W1, B1, actw, lane);
  v8f acc = mlp_layer_out<2>(actw, 64, W2, lane);
  // column sum over the 16 element-rows: 8 VGPRs + cross-half shuffle
  float part = 0.f;
#pragma unroll
  for (int r = 0; r < 8; ++r) part += acc[r];
  part += __shfl_xor(part, 16);
  int n = lane & 15;
  part += 16.0f * B2[n];   // bias added per element then summed over 16
  if (lane < 16) sdst[n] = (half_t)part;
}

// ---------------------------------------------------------------------------
// Main kernel: 256 threads (8 wave32), 16 samples per workgroup.
// ---------------------------------------------------------------------------
__global__ __launch_bounds__(256)
void barrier_net_kernel(const float* __restrict__ x, float* __restrict__ out,
                        const half_t* __restrict__ wt,
                        const float* __restrict__ bs) {
  __shared__ __align__(16) half_t actbuf[8][16 * 64];  // per-wave activation staging
  __shared__ __align__(16) half_t staged[3 * 16 * 32]; // sN | sO | psi inputs
  __shared__ float hb[16 * 2];  // psi head outputs
  __shared__ float fb[16 * 2];  // analytic barrier forces
  half_t* sbufN = staged;
  half_t* sbufO = staged + 512;
  half_t* psib  = staged + 1024;

  const int tid = threadIdx.x, w = tid >> 5, lane = tid & 31;
  const int sampleBase = blockIdx.x * 16;

  if (tid < 192) ((float4*)staged)[tid] = make_float4(0.f, 0.f, 0.f, 0.f);
  __syncthreads();

  half_t* actw = actbuf[w];

  // --- phase 1: per-element deep-set encoders ------------------------------
  // 4 pipelines per wave (2 samples x {neighbors, obstacles}); keep the loop
  // ROLLED so only one pipeline body exists -> low VGPR pressure, good
  // wave residency per SIMD32 while WMMA/LDS latencies are in flight.
  const half_t* W0s[2] = {wt + W_PHIN0, wt + W_PHIO0};
  const half_t* W1s[2] = {wt + W_PHIN1, wt + W_PHIO1};
  const half_t* W2s[2] = {wt + W_PHIN2, wt + W_PHIO2};
  const float*  B0s[2] = {bs + B_PHIN0, bs + B_PHIO0};
  const float*  B1s[2] = {bs + B_PHIN1, bs + B_PHIO1};
  const float*  B2s[2] = {bs + B_PHIN2, bs + B_PHIO2};
  half_t*       sds[2] = {sbufN, sbufO};
#pragma unroll 1
  for (int p = 0; p < 4; ++p) {
    int ob = p & 1;                 // wave-uniform
    int sl = w * 2 + (p >> 1);
    const float* xr = x + (size_t)(sampleBase + sl) * 101;
    phi_pipe(xr, ob, W0s[ob], B0s[ob], W1s[ob], B1s[ob], W2s[ob], B2s[ob],
             actw, sds[ob] + sl * 32, lane);
  }
  __syncthreads();

  // --- phase 2: rho heads (M=16 samples) + analytic terms, in parallel ----
  if (w == 0) {
    mlp_layer64<1>(sbufN, 32, wt + W_RHON0, bs + B_RHON0, actw, lane);
    v8f acc = mlp_layer_out<2>(actw, 64, wt + W_RHON1, lane);
    int n = lane & 15, mh = (lane >> 4) << 3;
    if (n < 8) {
      float bv = bs[B_RHON1 + n];
#pragma unroll
      for (int r = 0; r < 8; ++r) psib[(r + mh) * 32 + n] = (half_t)(acc[r] + bv);
    }
  } else if (w == 1) {
    mlp_layer64<1>(sbufO, 32, wt + W_RHOO0, bs + B_RHOO0, actw, lane);
    v8f acc = mlp_layer_out<2>(actw, 64, wt + W_RHOO1, lane);
    int n = lane & 15, mh = (lane >> 4) << 3;
    if (n < 8) {
      float bv = bs[B_RHOO1 + n];
#pragma unroll
      for (int r = 0; r < 8; ++r) psib[(r + mh) * 32 + 8 + n] = (half_t)(acc[r] + bv);
    }
  } else if (w == 2) {
    if (lane < 16) {   // goal features g = x[:,1:3]
      const float* xr = x + (size_t)(sampleBase + lane) * 101;
      psib[lane * 32 + 16] = (half_t)xr[1];
      psib[lane * 32 + 17] = (half_t)xr[2];
    }
  } else if (w == 3) {
    if (lane < 16) {   // analytic barrier forces f_n + f_o
      const float* xr = x + (size_t)(sampleBase + lane) * 101;
      float fx = 0.f, fy = 0.f;
#pragma unroll
      for (int j = 0; j < 16; ++j) {
        float px = -xr[5 + 4 * j], py = -xr[6 + 4 * j];
        float dist = sqrtf(px * px + py * py);
        float hbv  = dist - 0.49f;            // D_ROBOT
        float coef = 0.05f / hbv;             // B_GAMMA*sign(hb)/|hb|^1
        fx += coef * px; fy += coef * py;
      }
      float pox = xr[69], poy = xr[70];
      float dn = sqrtf(pox * pox + poy * poy);
      float d1 = 0.5f * dn / fmaxf(fabsf(pox), fabsf(poy));  // R_OBS
      float ho = dn - d1 - 0.15f;                            // R_AGENT
      float co = 0.05f / ho;
      fb[lane * 2]     = fx + 16.0f * co * pox;              // NO * co * po
      fb[lane * 2 + 1] = fy + 16.0f * co * poy;
    }
  }
  __syncthreads();

  // --- phase 3: psi head (M=16 samples) ------------------------------------
  if (w == 0) {
    mlp_layer64<1>(psib, 32, wt + W_PSI0, bs + B_PSI0, actw, lane);
    mlp_layer64<2>(actw, 64, wt + W_PSI1, bs + B_PSI1, actw, lane);
    v8f acc = mlp_layer_out<2>(actw, 64, wt + W_PSI2, lane);
    int n = lane & 15, mh = (lane >> 4) << 3;
    if (n < 2) {
      float bv = bs[B_PSI2 + n];
#pragma unroll
      for (int r = 0; r < 8; ++r) hb[(r + mh) * 2 + n] = acc[r] + bv;
    }
  }
  __syncthreads();

  // --- phase 4: combine + tanh squash, write output -------------------------
  if (tid < 16) {
    int s = sampleBase + tid;
    // empty = (tanh(h)+1)/2 * 4 - 2 = 2*tanh(h); same affine for final action
    float ax = 2.0f * tanhf(hb[tid * 2])     + fb[tid * 2];
    float ay = 2.0f * tanhf(hb[tid * 2 + 1]) + fb[tid * 2 + 1];
    out[(size_t)s * 2]     = 2.0f * tanhf(ax);
    out[(size_t)s * 2 + 1] = 2.0f * tanhf(ay);
  }
}

// ---------------------------------------------------------------------------
extern "C" void kernel_launch(void* const* d_in, const int* in_sizes, int n_in,
                              void* d_out, int out_size, void* d_ws, size_t ws_size,
                              hipStream_t stream) {
  const float* x = (const float*)d_in[0];
  half_t* wbase = (half_t*)d_ws;
  float*  bbase = (float*)((char*)d_ws + BIAS_BYTE_OFF);

  struct L { int wi, bi, din, dout, woff, boff; };
  static const L Ls[13] = {
    { 1,  2,  4, 64, W_PHIN0, B_PHIN0}, { 3,  4, 64, 64, W_PHIN1, B_PHIN1},
    { 5,  6, 64, 16, W_PHIN2, B_PHIN2},
    { 7,  8, 16, 64, W_RHON0, B_RHON0}, { 9, 10, 64,  8, W_RHON1, B_RHON1},
    {11, 12,  2, 64, W_PHIO0, B_PHIO0}, {13, 14, 64, 64, W_PHIO1, B_PHIO1},
    {15, 16, 64, 16, W_PHIO2, B_PHIO2},
    {17, 18, 16, 64, W_RHOO0, B_RHOO0}, {19, 20, 64,  8, W_RHOO1, B_RHOO1},
    {21, 22, 18, 64, W_PSI0,  B_PSI0 }, {23, 24, 64, 64, W_PSI1,  B_PSI1 },
    {25, 26, 64,  2, W_PSI2,  B_PSI2 },
  };
  for (int i = 0; i < 13; ++i) {
    const L& l = Ls[i];
    int KT = (l.din + 31) / 32, NT = (l.dout + 15) / 16;
    int total = KT * NT * 512;
    swizzle_w_kernel<<<(total + 255) / 256, 256, 0, stream>>>(
        (const float*)d_in[l.wi], l.din, l.dout, wbase + l.woff, KT, NT);
    int npad = NT * 16;
    pad_bias_kernel<<<1, npad, 0, stream>>>(
        (const float*)d_in[l.bi], l.dout, bbase + l.boff, npad);
  }

  int nd = in_sizes[0] / 101;           // 65536
  int grid = nd / 16;                   // 16 samples per workgroup
  barrier_net_kernel<<<grid, 256, 0, stream>>>(x, (float*)d_out, wbase, bbase);
}
